// ImprovedSNN_15384572854784
// MI455X (gfx1250) — matrix operand order repaired and checked by reading, not compiled, not run on hardware
//
#include <hip/hip_runtime.h>

typedef __attribute__((ext_vector_type(16))) _Float16 v16h;
typedef __attribute__((ext_vector_type(8)))  _Float16 v8h;
typedef __attribute__((ext_vector_type(8)))  float    v8f;
typedef __attribute__((ext_vector_type(4)))  float    v4f;

#define B_TOT   4096
#define F_IN    256
#define T_STEPS 100
#define NTG     (T_STEPS/4)
#define H1      128
#define H2      64
#define H3      32
#define BTILE   32
#define NTHR    256

// LDS layout (in _Float16 units unless noted)
#define W1H_N 32768            // 8 ktiles * 8 ntiles * 512
#define W2H_N 8192             // 4 * 4 * 512
#define W3H_N 2048             // 2 * 2 * 512
#define XLD   264              // 256 + 8 pad
#define S1LD  136
#define S2LD  72
#define S3LD  40
#define XSTEP (BTILE*XLD)
#define X4_N  (4*XSTEP)

#define HALF_N (W1H_N + W2H_N + W3H_N + 2*X4_N + BTILE*(S1LD + S2LD + S3LD))
#define SMEM_BYTES (HALF_N*2 + (H1 + H2 + H3 + 64 + 2)*4)

__device__ __forceinline__ v16h load_afrag(const _Float16* p) {
  // A 16x32 f16 fragment: lane<16 -> K = 0..15 of its row (+16 for lanes 16..31)
  v8h lo = *(const v8h*)p;
  v8h hi = *(const v8h*)(p + 8);
  v16h a;
#pragma unroll
  for (int j = 0; j < 8; ++j) { a[j] = lo[j]; a[j + 8] = hi[j]; }
  return a;
}

__device__ __forceinline__ v16h load_bfrag(const _Float16* p) {
  // pre-swizzled: lane's 16 contiguous halfs
  v8h lo = *(const v8h*)p;
  v8h hi = *(const v8h*)(p + 8);
  v16h b;
#pragma unroll
  for (int j = 0; j < 8; ++j) { b[j] = lo[j]; b[j + 8] = hi[j]; }
  return b;
}

// LIF update on a 16x16 f32 accumulator tile held in WMMA C layout.
// dt*tau_mem_inv = 0.1, dt*tau_syn_inv = 0.2 -> v' = 0.9v + 0.1i ; i' = 0.8i + input
__device__ __forceinline__ void lif_store(const v8f& acc, float cadd,
                                          v8f& v, v8f& i,
                                          _Float16* S, int ld, int mt, int nt, int lane) {
  const int mbase = mt*16 + ((lane >> 4) << 3);
  const int n     = nt*16 + (lane & 15);
#pragma unroll
  for (int r = 0; r < 8; ++r) {
    float y  = acc[r] + cadd;
    float vd = 0.9f*v[r] + 0.1f*i[r];
    float z  = (vd > 1.0f) ? 1.0f : 0.0f;
    v[r] = (vd > 1.0f) ? 0.0f : vd;
    i[r] = 0.8f*i[r] + y;
    S[(mbase + r)*ld + n] = (_Float16)z;
  }
}

__global__ __launch_bounds__(NTHR) void snn_fused(
    const float* __restrict__ spikes,
    const float* __restrict__ g0,  const float* __restrict__ bb0,
    const float* __restrict__ m0,  const float* __restrict__ vv0,
    const float* __restrict__ W1,  const float* __restrict__ b1,
    const float* __restrict__ g1,  const float* __restrict__ bb1,
    const float* __restrict__ m1,  const float* __restrict__ vv1,
    const float* __restrict__ W2,  const float* __restrict__ b2,
    const float* __restrict__ g2,  const float* __restrict__ bb2,
    const float* __restrict__ m2,  const float* __restrict__ vv2,
    const float* __restrict__ W3,  const float* __restrict__ b3,
    const float* __restrict__ g3,  const float* __restrict__ bb3,
    const float* __restrict__ m3,  const float* __restrict__ vv3,
    const float* __restrict__ Wout, const float* __restrict__ bout,
    float* __restrict__ out)
{
  extern __shared__ char smem_raw[];
  _Float16* sW1 = (_Float16*)smem_raw;
  _Float16* sW2 = sW1 + W1H_N;
  _Float16* sW3 = sW2 + W2H_N;
  _Float16* sX0 = sW3 + W3H_N;        // double-buffered spike tiles; also init scratch
  _Float16* sX1 = sX0 + X4_N;
  _Float16* sS1 = sX1 + X4_N;
  _Float16* sS2 = sS1 + BTILE*S1LD;
  _Float16* sS3 = sS2 + BTILE*S2LD;
  float* sC1 = (float*)(sS3 + BTILE*S3LD);
  float* sC2 = sC1 + H1;
  float* sC3 = sC2 + H2;
  float* sWo = sC3 + H3;   // 2x32
  float* sbo = sWo + 64;   // 2

  const int tid  = threadIdx.x;
  const int lane = tid & 31;
  const int wv   = tid >> 5;
  const int b0   = blockIdx.x * BTILE;

  // ---- init: fold BN0/1/2/3 into f16 fragment-swizzled weights + per-channel constants ----
  {
    float* tf   = (float*)sX0;             // scratch (overwritten by spike tiles later)
    float* tsc0 = tf;                      // 256
    float* tsh0 = tf + 256;                // 256
    float* tsc1 = tf + 512;                // 128
    float* tsc2 = tf + 640;                // 64
    float* tsc3 = tf + 704;                // 32
    {
      float sc = g0[tid] * rsqrtf(vv0[tid] + 1e-5f);
      tsc0[tid] = sc;
      tsh0[tid] = bb0[tid] - m0[tid]*sc;
    }
    if (tid < H1) tsc1[tid] = g1[tid] * rsqrtf(vv1[tid] + 1e-5f);
    if (tid < H2) tsc2[tid] = g2[tid] * rsqrtf(vv2[tid] + 1e-5f);
    if (tid < H3) tsc3[tid] = g3[tid] * rsqrtf(vv3[tid] + 1e-5f);
    if (tid < 64) sWo[tid] = Wout[tid];
    if (tid < 2)  sbo[tid] = bout[tid];
    __syncthreads();

    if (tid < H1) {
      float base = b1[tid];
      for (int k = 0; k < F_IN; ++k) base += W1[tid*F_IN + k] * tsh0[k];
      sC1[tid] = (base - m1[tid]) * tsc1[tid] + bb1[tid];
    }
    if (tid < H2) sC2[tid] = (b2[tid] - m2[tid]) * tsc2[tid] + bb2[tid];
    if (tid < H3) sC3[tid] = (b3[tid] - m3[tid]) * tsc3[tid] + bb3[tid];

    // B-fragment-swizzled weights: frag f = nt*KT + kt; lane l owns
    // n = nt*16+(l&15), K = kt*32 + ((l>>4)<<4) + e, e=0..15 contiguous.
    for (int idx = tid; idx < W1H_N; idx += NTHR) {
      int f = idx >> 9, r = idx & 511, l = r >> 4, e = r & 15;
      int kt = f & 7, nt = f >> 3;
      int n = nt*16 + (l & 15);
      int k = kt*32 + ((l >> 4) << 4) + e;
      sW1[idx] = (_Float16)(W1[n*F_IN + k] * tsc0[k] * tsc1[n]);
    }
    for (int idx = tid; idx < W2H_N; idx += NTHR) {
      int f = idx >> 9, r = idx & 511, l = r >> 4, e = r & 15;
      int kt = f & 3, nt = f >> 2;
      int n = nt*16 + (l & 15);
      int k = kt*32 + ((l >> 4) << 4) + e;
      sW2[idx] = (_Float16)(W2[n*H1 + k] * tsc2[n]);
    }
    for (int idx = tid; idx < W3H_N; idx += NTHR) {
      int f = idx >> 9, r = idx & 511, l = r >> 4, e = r & 15;
      int kt = f & 1, nt = f >> 1;
      int n = nt*16 + (l & 15);
      int k = kt*32 + ((l >> 4) << 4) + e;
      sW3[idx] = (_Float16)(W3[n*H2 + k] * tsc3[n]);
    }
  }
  __syncthreads();

  // ---- per-wave tile assignment ----
  const int mt1 = wv >> 2;            // L1: [32x128] = 2x8 tiles, 2 per wave
  const int ntA = (wv & 3) << 1;
  const int ntB = ntA | 1;
  const int mt2 = wv >> 2;            // L2: [32x64] = 2x4 tiles, 1 per wave
  const int nt2 = wv & 3;
  const int mt3 = wv >> 1;            // L3: [32x32] = 2x2 tiles, waves 0-3
  const int nt3 = wv & 1;

  const float cA  = sC1[ntA*16 + (lane & 15)];
  const float cB  = sC1[ntB*16 + (lane & 15)];
  const float c2v = sC2[nt2*16 + (lane & 15)];
  const float c3v = (wv < 4) ? sC3[nt3*16 + (lane & 15)] : 0.0f;

  // persistent LIF state in WMMA C layout
  v8f v1a = {}, i1a = {}, v1b = {}, i1b = {};
  v8f v2  = {}, i2  = {};
  v8f v3  = {}, i3  = {};
  float vo = 0.0f, io = 0.0f, accO = 0.0f;   // output layer (threads 0..63)

  // per-thread spike base: feature = tid (NTHR == F_IN), t contiguous
  const float* sp = spikes + (size_t)b0*(F_IN*T_STEPS) + (size_t)tid*T_STEPS;

  // ---- prologue: stage group 0 into buffer 0 ----
  for (int b = 0; b < BTILE; ++b) {
    v4f s4 = *(const v4f*)(sp + (size_t)b*(F_IN*T_STEPS));
    int xb = b*XLD + tid;
    sX0[xb]           = (_Float16)s4[0];
    sX0[XSTEP   + xb] = (_Float16)s4[1];
    sX0[2*XSTEP + xb] = (_Float16)s4[2];
    sX0[3*XSTEP + xb] = (_Float16)s4[3];
  }
  __syncthreads();

  for (int tg = 0; tg < NTG; ++tg) {
    const _Float16* Xc = (tg & 1) ? sX1 : sX0;
    _Float16*       Xn = (tg & 1) ? sX0 : sX1;
    const bool more  = (tg + 1 < NTG);
    const int  tnext = (tg + 1) * 4;

    for (int tt = 0; tt < 4; ++tt) {
      // software pipeline: issue next group's global loads (8 of 32 rows per tt)
      v4f pre[8];
      if (more) {
#pragma unroll
        for (int j = 0; j < 8; ++j) {
          pre[j] = *(const v4f*)(sp + (size_t)(tt*8 + j)*(F_IN*T_STEPS) + tnext);
        }
      }

      const _Float16* X = Xc + tt*XSTEP;
      // ---- layer 1: [32x256] x [256x128] ----
      {
        v8f aca = {}, acb = {};
        const _Float16* rowp = X + (mt1*16 + (lane & 15))*XLD + ((lane >> 4) << 4);
#pragma unroll
        for (int kt = 0; kt < 8; ++kt) {
          v16h a  = load_afrag(rowp + kt*32);
          v16h bA = load_bfrag(sW1 + (ntA*8 + kt)*512 + lane*16);
          v16h bB = load_bfrag(sW1 + (ntB*8 + kt)*512 + lane*16);
          aca = __builtin_amdgcn_wmma_f32_16x16x32_f16(false, a, false, bA, (short)0, aca, false, false);
          acb = __builtin_amdgcn_wmma_f32_16x16x32_f16(false, a, false, bB, (short)0, acb, false, false);
        }
        lif_store(aca, cA, v1a, i1a, sS1, S1LD, mt1, ntA, lane);
        lif_store(acb, cB, v1b, i1b, sS1, S1LD, mt1, ntB, lane);
      }
      __syncthreads();

      // ---- layer 2: [32x128] x [128x64] ----
      {
        v8f acc = {};
        const _Float16* rowp = sS1 + (mt2*16 + (lane & 15))*S1LD + ((lane >> 4) << 4);
#pragma unroll
        for (int kt = 0; kt < 4; ++kt) {
          v16h a = load_afrag(rowp + kt*32);
          v16h b = load_bfrag(sW2 + (nt2*4 + kt)*512 + lane*16);
          acc = __builtin_amdgcn_wmma_f32_16x16x32_f16(false, a, false, b, (short)0, acc, false, false);
        }
        lif_store(acc, c2v, v2, i2, sS2, S2LD, mt2, nt2, lane);
      }
      __syncthreads();

      // ---- layer 3: [32x64] x [64x32] (waves 0-3) ----
      if (wv < 4) {
        v8f acc = {};
        const _Float16* rowp = sS2 + (mt3*16 + (lane & 15))*S2LD + ((lane >> 4) << 4);
#pragma unroll
        for (int kt = 0; kt < 2; ++kt) {
          v16h a = load_afrag(rowp + kt*32);
          v16h b = load_bfrag(sW3 + (nt3*2 + kt)*512 + lane*16);
          acc = __builtin_amdgcn_wmma_f32_16x16x32_f16(false, a, false, b, (short)0, acc, false, false);
        }
        lif_store(acc, c3v, v3, i3, sS3, S3LD, mt3, nt3, lane);
      }

      // drain prefetched rows into the other X buffer (hides load latency in the
      // L3 -> output sync gap; consumed only after the end-of-tt barrier of tt=3)
      if (more) {
#pragma unroll
        for (int j = 0; j < 8; ++j) {
          int xb = (tt*8 + j)*XLD + tid;
          Xn[xb]           = (_Float16)pre[j][0];
          Xn[XSTEP   + xb] = (_Float16)pre[j][1];
          Xn[2*XSTEP + xb] = (_Float16)pre[j][2];
          Xn[3*XSTEP + xb] = (_Float16)pre[j][3];
        }
      }
      __syncthreads();

      // ---- output layer: [32x32] x [32x2], scalar VALU (threads 0..63) ----
      if (tid < 64) {
        const int bb = tid >> 1, o = tid & 1;
        float y = sbo[o];
        const _Float16* s3r = sS3 + bb*S3LD;
        const float* wr = sWo + o*32;
#pragma unroll
        for (int k = 0; k < 32; ++k) y += (float)s3r[k] * wr[k];
        float vd = 0.9f*vo + 0.1f*io;
        float z  = (vd > 1.0f) ? 1.0f : 0.0f;
        vo = (vd > 1.0f) ? 0.0f : vd;
        io = 0.8f*io + y;
        accO += z;
      }
      __syncthreads();
    }
  }

  if (tid < 64) out[(size_t)(b0 + (tid >> 1))*2 + (tid & 1)] = accO;
}

extern "C" void kernel_launch(void* const* d_in, const int* in_sizes, int n_in,
                              void* d_out, int out_size, void* d_ws, size_t ws_size,
                              hipStream_t stream) {
  (void)in_sizes; (void)n_in; (void)d_ws; (void)ws_size; (void)out_size;
  const float* spikes = (const float*)d_in[0];
  const float* g0  = (const float*)d_in[1];
  const float* bb0 = (const float*)d_in[2];
  const float* m0  = (const float*)d_in[3];
  const float* vv0 = (const float*)d_in[4];
  const float* W1  = (const float*)d_in[5];
  const float* b1  = (const float*)d_in[6];
  const float* g1  = (const float*)d_in[7];
  const float* bb1 = (const float*)d_in[8];
  const float* m1  = (const float*)d_in[9];
  const float* vv1 = (const float*)d_in[10];
  const float* W2  = (const float*)d_in[11];
  const float* b2  = (const float*)d_in[12];
  const float* g2  = (const float*)d_in[13];
  const float* bb2 = (const float*)d_in[14];
  const float* m2  = (const float*)d_in[15];
  const float* vv2 = (const float*)d_in[16];
  const float* W3  = (const float*)d_in[17];
  const float* b3  = (const float*)d_in[18];
  const float* g3  = (const float*)d_in[19];
  const float* bb3 = (const float*)d_in[20];
  const float* m3  = (const float*)d_in[21];
  const float* vv3 = (const float*)d_in[22];
  const float* Wout = (const float*)d_in[23];
  const float* bout = (const float*)d_in[24];

  dim3 grid(B_TOT / BTILE);   // 128 blocks
  dim3 block(NTHR);           // 8 wave32
  snn_fused<<<grid, block, SMEM_BYTES, stream>>>(
      spikes,
      g0, bb0, m0, vv0,
      W1, b1, g1, bb1, m1, vv1,
      W2, b2, g2, bb2, m2, vv2,
      W3, b3, g3, bb3, m3, vv3,
      Wout, bout,
      (float*)d_out);
}
